// FusedConv_42099269436001
// MI455X (gfx1250) — matrix operand order repaired and checked by reading, not compiled, not run on hardware
//
#include <hip/hip_runtime.h>
#include <hip/hip_bf16.h>

typedef __bf16 bf16x16 __attribute__((ext_vector_type(16)));
typedef float  f32x8   __attribute__((ext_vector_type(8)));

union ABOp {
  bf16x16 v;
  unsigned short us[16];
  uint4 q[2];
};

static __device__ __forceinline__ unsigned short f2bf(float f) {
  union { float f; unsigned u; } x; x.f = f;
  unsigned r = x.u + 0x7FFFu + ((x.u >> 16) & 1u);
  return (unsigned short)(r >> 16);
}

static __device__ __forceinline__ float gelu_exact(float x) {
  return 0.5f * x * (1.0f + erff(x * 0.70710678118654752f));
}

// Transpose + f32->bf16 convert: dst[n*K + k] = bf16(src[k*srcLd + colOff + n])
__global__ void prep_transpose_bf16(const float* __restrict__ src,
                                    unsigned short* __restrict__ dst,
                                    int K, int N, int srcLd, int colOff) {
  int idx = blockIdx.x * blockDim.x + threadIdx.x;
  if (idx >= K * N) return;
  int n = idx / K;
  int k = idx - n * K;
  dst[n * K + k] = f2bf(src[k * srcLd + colOff + n]);
}

#define WAVES_PER_WG 4
#define LDS_PER_WAVE 20480   // 8192 (h bf16) + 9216 (tmp f32) + 3072 (rw tile f32)

__global__ __launch_bounds__(128)
void fused_equiv_mlp_kernel(const float* __restrict__ edges,     // [E][32]
                            const float* __restrict__ feats,     // [E][16][3]
                            const float* __restrict__ basis,     // [E][3][9]
                            const unsigned short* __restrict__ Wt1, // [256][32]  bf16
                            const unsigned short* __restrict__ Wt2, // [256][256] bf16
                            const unsigned short* __restrict__ Wt3, // [768][256] bf16
                            const float* __restrict__ b1,
                            const float* __restrict__ b2,
                            float* __restrict__ out)              // [E][16][3]
{
  extern __shared__ char smem_raw[];
  const int wave = threadIdx.x >> 5;
  const int lane = threadIdx.x & 31;
  const int hl   = lane >> 4;    // which 16-lane half
  const int ln   = lane & 15;

  char* wbase = smem_raw + wave * LDS_PER_WAVE;
  unsigned short* sh = (unsigned short*)wbase;           // h: [16 rows][256] bf16
  float* stmp = (float*)(wbase + 8192);                  // tmp: [16 e][48 j][3 d] f32
  float* srw  = (float*)(wbase + 8192 + 9216);           // rw tile: [16 e][48 j] f32

  const int e0 = (blockIdx.x * WAVES_PER_WG + wave) * 16;

  // ---------- phase 0: tmp[e][j][d] = sum_d' f[e][i][d'] * basis[e][d'][nf*3+d], j=i*3+nf
  {
    const int e = ln;
    const float* bg = basis + (size_t)(e0 + e) * 27;
    float bas[27];
    #pragma unroll
    for (int i = 0; i < 27; ++i) bas[i] = bg[i];
    #pragma unroll
    for (int ii = 0; ii < 8; ++ii) {
      int i = hl * 8 + ii;
      const float* fg = feats + ((size_t)(e0 + e) * 16 + i) * 3;
      float f0 = fg[0], f1 = fg[1], f2 = fg[2];
      #pragma unroll
      for (int nf = 0; nf < 3; ++nf) {
        #pragma unroll
        for (int d = 0; d < 3; ++d) {
          float t = f0 * bas[0 * 9 + nf * 3 + d]
                  + f1 * bas[1 * 9 + nf * 3 + d]
                  + f2 * bas[2 * 9 + nf * 3 + d];
          stmp[(e * 48 + (i * 3 + nf)) * 3 + d] = t;
        }
      }
    }
  }

  // ---------- layer 1: h1 = gelu(edges @ W1 + b1), K = 32 (single WMMA K-chunk)
  ABOp a1;
  {
    const float* eg = edges + (size_t)(e0 + ln) * 32;
    // A layout (16-bit, 16x32): lanes0-15 hold K = 0..7 / 16..23; lanes16-31 K = 8..15 / 24..31
    int kb = hl * 8;
    float4 x0 = *(const float4*)(eg + kb);
    float4 x1 = *(const float4*)(eg + kb + 4);
    float4 x2 = *(const float4*)(eg + 16 + kb);
    float4 x3 = *(const float4*)(eg + 16 + kb + 4);
    a1.us[0] = f2bf(x0.x);  a1.us[1] = f2bf(x0.y);  a1.us[2] = f2bf(x0.z);  a1.us[3] = f2bf(x0.w);
    a1.us[4] = f2bf(x1.x);  a1.us[5] = f2bf(x1.y);  a1.us[6] = f2bf(x1.z);  a1.us[7] = f2bf(x1.w);
    a1.us[8] = f2bf(x2.x);  a1.us[9] = f2bf(x2.y);  a1.us[10] = f2bf(x2.z); a1.us[11] = f2bf(x2.w);
    a1.us[12] = f2bf(x3.x); a1.us[13] = f2bf(x3.y); a1.us[14] = f2bf(x3.z); a1.us[15] = f2bf(x3.w);
  }
  #pragma unroll
  for (int nt = 0; nt < 16; ++nt) {
    ABOp b;
    const unsigned short* wp = Wt1 + (size_t)(nt * 16 + ln) * 32 + hl * 8;
    b.q[0] = *(const uint4*)(wp);
    b.q[1] = *(const uint4*)(wp + 16);
    f32x8 c = {0.f, 0.f, 0.f, 0.f, 0.f, 0.f, 0.f, 0.f};
    c = __builtin_amdgcn_wmma_f32_16x16x32_bf16(false, a1.v, false, b.v, (short)0, c, false, false);
    float bias = b1[nt * 16 + ln];
    #pragma unroll
    for (int r = 0; r < 8; ++r)
      sh[(r + 8 * hl) * 256 + nt * 16 + ln] = f2bf(gelu_exact(c[r] + bias));
  }

  // ---------- layer 2: h2 = gelu(h1 @ W2 + b2); all 16 C tiles live in registers
  f32x8 c2[16];
  #pragma unroll
  for (int nt = 0; nt < 16; ++nt)
    c2[nt] = (f32x8){0.f, 0.f, 0.f, 0.f, 0.f, 0.f, 0.f, 0.f};
  #pragma unroll
  for (int kc = 0; kc < 8; ++kc) {
    ABOp a;
    const unsigned short* hp = sh + ln * 256 + kc * 32 + hl * 8;
    a.q[0] = *(const uint4*)(hp);
    a.q[1] = *(const uint4*)(hp + 16);
    #pragma unroll
    for (int nt = 0; nt < 16; ++nt) {
      ABOp b;
      const unsigned short* wp = Wt2 + (size_t)(nt * 16 + ln) * 256 + kc * 32 + hl * 8;
      b.q[0] = *(const uint4*)(wp);
      b.q[1] = *(const uint4*)(wp + 16);
      c2[nt] = __builtin_amdgcn_wmma_f32_16x16x32_bf16(false, a.v, false, b.v, (short)0, c2[nt], false, false);
    }
  }
  #pragma unroll
  for (int nt = 0; nt < 16; ++nt) {
    float bias = b2[nt * 16 + ln];
    #pragma unroll
    for (int r = 0; r < 8; ++r)
      sh[(r + 8 * hl) * 256 + nt * 16 + ln] = f2bf(gelu_exact(c2[nt][r] + bias));
  }

  // ---------- layer 3 + per-edge contraction, one output head o at a time
  // Cache all 8 A chunks of h2 (64 VGPRs), reused across all 16 o's.
  ABOp a3[8];
  #pragma unroll
  for (int kc = 0; kc < 8; ++kc) {
    const unsigned short* hp = sh + ln * 256 + kc * 32 + hl * 8;
    a3[kc].q[0] = *(const uint4*)(hp);
    a3[kc].q[1] = *(const uint4*)(hp + 16);
  }

  for (int o = 0; o < 16; ++o) {
    f32x8 c3[3];
    #pragma unroll
    for (int tt = 0; tt < 3; ++tt)
      c3[tt] = (f32x8){0.f, 0.f, 0.f, 0.f, 0.f, 0.f, 0.f, 0.f};
    #pragma unroll
    for (int kc = 0; kc < 8; ++kc) {
      #pragma unroll
      for (int tt = 0; tt < 3; ++tt) {
        ABOp b;
        const unsigned short* wp =
            Wt3 + (size_t)(o * 48 + tt * 16 + ln) * 256 + kc * 32 + hl * 8;
        b.q[0] = *(const uint4*)(wp);
        b.q[1] = *(const uint4*)(wp + 16);
        c3[tt] = __builtin_amdgcn_wmma_f32_16x16x32_bf16(false, a3[kc].v, false, b.v,
                                                         (short)0, c3[tt], false, false);
      }
    }
    // spill rw tile for this o: rw[e][j], j = tt*16 + col
    #pragma unroll
    for (int tt = 0; tt < 3; ++tt)
      #pragma unroll
      for (int r = 0; r < 8; ++r)
        srw[(r + 8 * hl) * 48 + tt * 16 + ln] = c3[tt][r];

    // out[e][o][d] = sum_j rw[e][j] * tmp[e][j][d]; halves split the j range
    {
      const int e = ln;
      float acc0 = 0.f, acc1 = 0.f, acc2 = 0.f;
      #pragma unroll
      for (int jj = 0; jj < 24; ++jj) {
        int j = hl * 24 + jj;
        float rv = srw[e * 48 + j];
        const float* tp = stmp + (e * 48 + j) * 3;
        acc0 += rv * tp[0];
        acc1 += rv * tp[1];
        acc2 += rv * tp[2];
      }
      acc0 += __shfl_xor(acc0, 16, 32);
      acc1 += __shfl_xor(acc1, 16, 32);
      acc2 += __shfl_xor(acc2, 16, 32);
      if (hl == 0) {
        float* op = out + (size_t)(e0 + e) * 48 + o * 3;
        op[0] = acc0; op[1] = acc1; op[2] = acc2;
      }
    }
  }
}

extern "C" void kernel_launch(void* const* d_in, const int* in_sizes, int n_in,
                              void* d_out, int out_size, void* d_ws, size_t ws_size,
                              hipStream_t stream) {
  const float* edges = (const float*)d_in[0];
  const float* feats = (const float*)d_in[1];
  const float* basis = (const float*)d_in[2];
  const float* W1 = (const float*)d_in[3];
  const float* b1 = (const float*)d_in[4];
  const float* W2 = (const float*)d_in[5];
  const float* b2 = (const float*)d_in[6];
  const float* W3 = (const float*)d_in[7];
  float* out = (float*)d_out;

  // Workspace: bf16 transposed weights ([N][K], contiguous K)
  unsigned short* Wt1 = (unsigned short*)d_ws;                              // 256*32  -> 16 KB
  unsigned short* Wt2 = (unsigned short*)((char*)d_ws + 16384);             // 256*256 -> 128 KB
  unsigned short* Wt3 = (unsigned short*)((char*)d_ws + 16384 + 131072);    // 768*256 -> 384 KB

  prep_transpose_bf16<<<(256 * 32 + 255) / 256, 256, 0, stream>>>(W1, Wt1, 32, 256, 256, 0);
  prep_transpose_bf16<<<(256 * 256 + 255) / 256, 256, 0, stream>>>(W2, Wt2, 256, 256, 256, 0);
  // Only the used slice of W3: columns [768, 1536)
  prep_transpose_bf16<<<(768 * 256 + 255) / 256, 256, 0, stream>>>(W3, Wt3, 256, 768, 1536, 768);

  const int E = 2 * 1024 * 32;                 // 65536 edges
  dim3 grid(E / (16 * WAVES_PER_WG));          // 1024 workgroups, 16 edges per wave
  fused_equiv_mlp_kernel<<<grid, 32 * WAVES_PER_WG, WAVES_PER_WG * LDS_PER_WAVE, stream>>>(
      edges, feats, basis, Wt1, Wt2, Wt3, b1, b2, out);
}
